// EarthAttention3D_59339268161782
// MI455X (gfx1250) — matrix operand order repaired and checked by reading, not compiled, not run on hardware
//
#include <hip/hip_runtime.h>

typedef __attribute__((ext_vector_type(16))) _Float16 v16h;
typedef __attribute__((ext_vector_type(8)))  _Float16 v8h;
typedef __attribute__((ext_vector_type(8)))  float    v8f;
typedef __attribute__((ext_vector_type(4)))  float    v4f;

#define NB   1024
#define NN   144
#define DDIM 192
#define NH   6
#define HD   32
#define SCALE 0.17677669529663687f   // 1/sqrt(32)

// LDS strides in elements; all row bases / fragment offsets are 16B aligned.
#define XS   200   // x    f16 [144][XS]        (row-major; A side)
#define W1S  200   // w1^T f16 [96][W1S]        (transposed; B side)
#define QKS  72    // q,k  f16 [144][QKS]       (row-major: q 0..31, k 32..63)
#define VTS  160   // v^T  f16 [32][VTS]        (transposed; B side, K padded to 160)
#define SSf  148   // scores f32 [144][SSf]
#define W2S  200   // w2^T f16 [192][W2S]       (transposed; overlays S region)
#define PS   160   // P    f16 [144][PS]        (row-major; A side, K padded to 160)
#define OS   200   // O    f16 [144][OS]        (row-major; A side)

#define OFF_XH   0                               // 57600
#define OFF_WT1  (OFF_XH  + NN*XS*2)             // +38400
#define OFF_QK   (OFF_WT1 + 96*W1S*2)            // +20736
#define OFF_VT   (OFF_QK  + NN*QKS*2)            // +10240
#define OFF_S    (OFF_VT  + HD*VTS*2)            // +85248 (>= 192*W2S*2 = 76800)
#define OFF_P    (OFF_S   + NN*SSf*4)            // +46080
#define OFF_OH   (OFF_P   + NN*PS*2)             // +57600
#define OFF_RED  (OFF_OH  + NN*OS*2)             // +576
#define SMEM_BYTES (OFF_RED + NN*4)              // = 316480 B < 320 KB

__device__ __forceinline__ v8f wmma32(const v16h& a, const v16h& b, const v8f& c) {
  return __builtin_amdgcn_wmma_f32_16x16x32_f16(false, a, false, b, (short)0, c,
                                                false, false);
}

// Fragment = two contiguous 16B LDS chunks at p+off and p+off+step.
__device__ __forceinline__ v16h frag2(const _Float16* p, int off, int step) {
  v8h lo = *(const v8h*)(p + off);
  v8h hi = *(const v8h*)(p + off + step);
  return __builtin_shufflevector(lo, hi, 0,1,2,3,4,5,6,7,8,9,10,11,12,13,14,15);
}
// A (16x32 f16, row-major source): K chunks {half*8.. , 16+half*8..}
__device__ __forceinline__ v16h fragA(const _Float16* row, int kb, int half) {
  return frag2(row, kb + half*8, 16);
}
// B (32x16 f16, transposed source row = fixed N): K chunks {half*8.. , 16+half*8..}
__device__ __forceinline__ v16h fragB(const _Float16* rowT, int kb, int half) {
  return frag2(rowT, kb + half*8, 16);
}

extern "C" __global__ __launch_bounds__(256)
void earth_attn_kernel(const float* __restrict__ x,
                       const float* __restrict__ w1,
                       const float* __restrict__ b1,
                       const float* __restrict__ w2,
                       const float* __restrict__ b2,
                       const float* __restrict__ bias_table,
                       const int*   __restrict__ pos_index,
                       const int*   __restrict__ mask,
                       float*       __restrict__ out)
{
  extern __shared__ char smem[];
  _Float16* XH  = (_Float16*)(smem + OFF_XH);
  _Float16* WT1 = (_Float16*)(smem + OFF_WT1);
  _Float16* QK  = (_Float16*)(smem + OFF_QK);
  _Float16* VT  = (_Float16*)(smem + OFF_VT);
  float*    S   = (float*)   (smem + OFF_S);
  _Float16* W2T = (_Float16*)(smem + OFF_S);    // phase-3 overlay
  _Float16* P16 = (_Float16*)(smem + OFF_P);
  _Float16* OH  = (_Float16*)(smem + OFF_OH);
  float*    RED = (float*)   (smem + OFF_RED);

  const int tid  = threadIdx.x;
  const int wave = tid >> 5;
  const int lane = tid & 31;
  const int half = lane >> 4;
  const int l16  = lane & 15;
  const int bb   = blockIdx.x;

  // ---- stage x_b (f32 -> f16); zero K-pad cols of V^T and P16 (144..159)
  const float* xb = x + (size_t)bb * (NN * DDIM);
  for (int idx = tid; idx < NN * DDIM; idx += 256) {
    int r = idx / DDIM, c = idx - r * DDIM;
    XH[r * XS + c] = (_Float16)xb[idx];
  }
  for (int idx = tid; idx < HD * 16; idx += 256)
    VT[(idx >> 4) * VTS + NN + (idx & 15)] = (_Float16)0.0f;
  for (int idx = tid; idx < NN * 16; idx += 256)
    P16[(idx >> 4) * PS + NN + (idx & 15)] = (_Float16)0.0f;
  __syncthreads();

  for (int h = 0; h < NH; ++h) {
    // ---- stage w1 head-slice TRANSPOSED: WT1[c][r] = w1[r][(c/32)*192 + h*32 + c%32]
    for (int idx = tid; idx < DDIM * 96; idx += 256) {
      int r = idx / 96, c = idx - r * 96;
      int gcol = (c >> 5) * DDIM + h * HD + (c & 31);
      WT1[c * W1S + r] = (_Float16)w1[r * (3 * DDIM) + gcol];
    }
    if (h + 1 < NH && lane == 0) {   // warm next head's w1 slice in L2
      __builtin_prefetch((const void*)&w1[(wave * 24) * (3 * DDIM) + (h + 1) * HD], 0, 0);
    }
    __syncthreads();

    // ---- phase 1: QKV_h = x @ w1_h + b1_h ; q,k row-major, v transposed
    for (int t = wave; t < 9 * 6; t += 8) {
      int tm = t / 6, tn = t - tm * 6;
      int c = tn * 16 + l16;                         // 0..95
      float bias1 = b1[(c >> 5) * DDIM + h * HD + (c & 31)];
      v8f acc;
      #pragma unroll
      for (int r = 0; r < 8; ++r) acc[r] = bias1;
      const _Float16* arow = XH + (tm * 16 + l16) * XS;
      const _Float16* brow = WT1 + c * W1S;
      #pragma unroll
      for (int kb = 0; kb < DDIM; kb += 32)
        acc = wmma32(fragA(arow, kb, half), fragB(brow, kb, half), acc);
      if (c < 64) {                                  // q,k: row-major scatter
        #pragma unroll
        for (int r = 0; r < 8; ++r)
          QK[(tm * 16 + r + half * 8) * QKS + c] = (_Float16)acc[r];
      } else {                                       // v: transposed, one packed store
        v8h pk;
        #pragma unroll
        for (int r = 0; r < 8; ++r) pk[r] = (_Float16)acc[r];
        *(v8h*)(VT + (c - 64) * VTS + tm * 16 + half * 8) = pk;
      }
    }
    __syncthreads();

    // ---- phase 2a: S = scale * q @ k^T + bias, masked (K=32 -> 1 WMMA/tile)
    for (int t = wave; t < 9 * 9; t += 8) {
      int tm = t / 9, tn = t - tm * 9;
      const _Float16* qrow = QK + (tm * 16 + l16) * QKS;        // q cols 0..31
      const _Float16* krow = QK + (tn * 16 + l16) * QKS + HD;   // k cols 32..63
      v16h av = fragA(qrow, 0, half);
      v16h bv = frag2(krow, half * 8, 16);   // K chunks {half*8.., 16+half*8..} within key row m
      v8f acc = {};
      acc = wmma32(av, bv, acc);
      int m = tn * 16 + l16;
      #pragma unroll
      for (int r = 0; r < 8; ++r) {
        int n = tm * 16 + r + half * 8;
        int pi = pos_index[n * NN + m];
        float val = acc[r] * SCALE + bias_table[pi * NH + h];
        if (mask[n * NN + m] == 0) val = -1.0e9f;
        S[n * SSf + m] = val;
      }
    }
    __syncthreads();

    // ---- phase 2b: row softmax; vectorized; write P as packed f16
    if (tid < NN) {
      const float* row = S + tid * SSf;
      _Float16* prow = P16 + tid * PS;
      float mx = -3.0e38f;
      for (int m0 = 0; m0 < NN; m0 += 8) {
        v4f a = *(const v4f*)(row + m0);
        v4f b = *(const v4f*)(row + m0 + 4);
        mx = fmaxf(mx, fmaxf(fmaxf(fmaxf(a[0], a[1]), fmaxf(a[2], a[3])),
                             fmaxf(fmaxf(b[0], b[1]), fmaxf(b[2], b[3]))));
      }
      float sum = 0.0f;
      for (int m0 = 0; m0 < NN; m0 += 8) {
        v4f a = *(const v4f*)(row + m0);
        v4f b = *(const v4f*)(row + m0 + 4);
        v8h pk;
        #pragma unroll
        for (int j = 0; j < 4; ++j) {
          float e = __expf(a[j] - mx); sum += e; pk[j] = (_Float16)e;
        }
        #pragma unroll
        for (int j = 0; j < 4; ++j) {
          float e = __expf(b[j] - mx); sum += e; pk[4 + j] = (_Float16)e;
        }
        *(v8h*)(prow + m0) = pk;
      }
      RED[tid] = 1.0f / sum;
    }
    __syncthreads();

    // ---- phase 2c: O[:, h*32+..] = (P @ v) * (1/rowsum); K padded to 160
    for (int t = wave; t < 9 * 2; t += 8) {
      int tm = t >> 1, tn = t & 1;
      v8f acc = {};
      const _Float16* arow = P16 + (tm * 16 + l16) * PS;
      const _Float16* brow = VT + (tn * 16 + l16) * VTS;  // v^T row j (pad zeroed)
      #pragma unroll
      for (int kb = 0; kb < 160; kb += 32)
        acc = wmma32(fragA(arow, kb, half), fragB(brow, kb, half), acc);
      #pragma unroll
      for (int r = 0; r < 8; ++r) {
        int n = tm * 16 + r + half * 8;
        OH[n * OS + h * HD + tn * 16 + l16] = (_Float16)(acc[r] * RED[n]);
      }
    }
    __syncthreads();
  }

  // ---- phase 3: out = O @ w2 + b2  (w2 staged transposed over the S region)
  for (int idx = tid; idx < DDIM * DDIM; idx += 256) {
    int r = idx / DDIM, c = idx - r * DDIM;
    W2T[c * W2S + r] = (_Float16)w2[idx];
  }
  __syncthreads();

  float* ob = out + (size_t)bb * (NN * DDIM);
  for (int t = wave; t < 9 * 12; t += 8) {
    int tm = t / 12, tn = t - tm * 12;
    int c = tn * 16 + l16;
    float bias2 = b2[c];
    v8f acc;
    #pragma unroll
    for (int r = 0; r < 8; ++r) acc[r] = bias2;
    const _Float16* arow = OH + (tm * 16 + l16) * OS;
    const _Float16* brow = W2T + c * W2S;
    #pragma unroll
    for (int kb = 0; kb < DDIM; kb += 32)
      acc = wmma32(fragA(arow, kb, half), fragB(brow, kb, half), acc);
    #pragma unroll
    for (int r = 0; r < 8; ++r) {
      int n = tm * 16 + r + half * 8;
      ob[(size_t)n * DDIM + c] = acc[r];
    }
  }
}

extern "C" void kernel_launch(void* const* d_in, const int* in_sizes, int n_in,
                              void* d_out, int out_size, void* d_ws, size_t ws_size,
                              hipStream_t stream) {
  const float* x  = (const float*)d_in[0];
  const float* w1 = (const float*)d_in[1];
  const float* b1 = (const float*)d_in[2];
  const float* w2 = (const float*)d_in[3];
  const float* b2 = (const float*)d_in[4];
  const float* bt = (const float*)d_in[5];
  const int*   pi = (const int*)d_in[6];
  const int*   mk = (const int*)d_in[7];
  float* outp = (float*)d_out;

  static_assert(SMEM_BYTES <= 320 * 1024, "exceeds gfx1250 WGP LDS");
  (void)hipFuncSetAttribute((const void*)earth_attn_kernel,
                            hipFuncAttributeMaxDynamicSharedMemorySize, SMEM_BYTES);

  earth_attn_kernel<<<NB, 256, SMEM_BYTES, stream>>>(x, w1, b1, w2, b2, bt, pi, mk, outp);
}
